// MultiHeadAttention_68341519614254
// MI455X (gfx1250) — compile-verified, matrix operand order checked by reading, not run on hardware
//
#include <hip/hip_runtime.h>
#include <cmath>

#define D_MODEL 768
#define NHEADS 12
#define HDIM 64

typedef __attribute__((ext_vector_type(16))) __bf16        v16bf;
typedef __attribute__((ext_vector_type(8)))  float         v8f;
typedef __attribute__((ext_vector_type(4)))  unsigned int  u32x4;

union Frag { u32x4 u[2]; v16bf v; };

// Hardware bf16 convert (RNE) -> lets clang use v_cvt_*bf16_f32 instead of
// a scalarized integer round-to-nearest-even sequence.
__device__ __forceinline__ unsigned short f2bf(float x) {
  __bf16 h = (__bf16)x;
  return __builtin_bit_cast(unsigned short, h);
}

__device__ __forceinline__ unsigned int pk2bf(float a, float b) {
  return (unsigned int)f2bf(a) | ((unsigned int)f2bf(b) << 16);
}

// CDNA5 async global->LDS copy, 16B per lane, tracked by ASYNCcnt.
// Generic pointers to LDS carry the LDS byte address in their low 32 bits
// (ISA 10.2 aperture rules), which is exactly what VDST wants.
__device__ __forceinline__ void async_g2l_b128(void* lds_dst, const void* gsrc) {
  unsigned ldsoff = (unsigned)(size_t)lds_dst;
  unsigned long long ga = (unsigned long long)(size_t)gsrc;
  asm volatile("global_load_async_to_lds_b128 %0, %1, off"
               :: "v"(ldsoff), "v"(ga)
               : "memory");
}
__device__ __forceinline__ void wait_async0() {
  asm volatile("s_wait_asynccnt 0x0" ::: "memory");
}

// Load one 16x32 bf16 WMMA fragment (A-layout; identical for B of an NT GEMM)
// from a row-major [16 x >=32] bf16 block with `stride` elements per row.
// lane<16: row=lane,     K chunks {0..7, 16..23}
// lane>=16: row=lane-16, K chunks {8..15, 24..31}
__device__ __forceinline__ v16bf ld_frag(const unsigned short* base, int stride) {
  const int lane = threadIdx.x & 31;
  const unsigned short* p = base + (lane & 15) * stride + (lane >> 4) * 8;
  Frag f;
  f.u[0] = *(const u32x4*)(p);
  f.u[1] = *(const u32x4*)(p + 16);
  return f.v;
}

__device__ __forceinline__ v8f wmma_bf16(v16bf a, v16bf b, v8f c) {
  return __builtin_amdgcn_wmma_f32_16x16x32_bf16(false, a, false, b, (short)0, c,
                                                 false, false);
}

// ---------------------------------------------------------------------------
// Weight transpose + fp32->bf16:  WT[o*n + i] = bf16(W[i*n + o])
// ---------------------------------------------------------------------------
__global__ void wt_cvt_t(const float* __restrict__ W,
                         unsigned short* __restrict__ WT, int n) {
  int idx = blockIdx.x * blockDim.x + threadIdx.x;
  if (idx >= n * n) return;
  int o = idx / n, i = idx - o * n;
  WT[idx] = f2bf(W[(size_t)i * n + o]);
}

// ---------------------------------------------------------------------------
// Generic NT GEMM:  C[M,N] = A[M,K] * Bt[N,K]^T (+bias) with layout modes.
//  a_is_fp32: A is fp32 (converted to bf16 while staging), else bf16
//             (staged with async global->LDS DMA).
//  mode 0: fp32 out [M,N]
//  mode 1: bf16 out, head-major       out[(h*S + m)*64 + d],  n=(h,d)
//  mode 2: bf16 out, head-transposed  out[(h*64 + d)*S + m]   (for V^T)
// Tile 128x128, K-step 32, 256 threads = 8 waves (2x4 subtiles of 16x16 each).
// ---------------------------------------------------------------------------
#define GT_M 128
#define GT_N 128
#define GT_K 32
#define SSTR 40   // padded LDS row stride (elems): 80B/row keeps b128 reads bank-spread

__global__ __launch_bounds__(256) void gemm_bf16_nt(
    const void* __restrict__ Aany, int a_is_fp32,
    const unsigned short* __restrict__ Bt,
    const float* __restrict__ bias,
    void* __restrict__ Cout,
    int M, int N, int Kd, int mode, float scale, int S)
{
  __shared__ __align__(16) unsigned short sA[GT_M * SSTR];
  __shared__ __align__(16) unsigned short sB[GT_N * SSTR];

  const int tid  = threadIdx.x;
  const int lane = tid & 31;
  const int wv   = tid >> 5;
  const int m0 = blockIdx.x * GT_M;
  const int n0 = blockIdx.y * GT_N;
  const int wm = (wv & 3) * 32;   // wave M offset within tile
  const int wn = (wv >> 2) * 64;  // wave N offset within tile

  const v8f vzero = {0.f,0.f,0.f,0.f,0.f,0.f,0.f,0.f};
  v8f acc[2][4];
  #pragma unroll
  for (int mt = 0; mt < 2; ++mt)
    #pragma unroll
    for (int nt = 0; nt < 4; ++nt) acc[mt][nt] = vzero;

  const int lrow  = tid >> 1;          // 0..127
  const int lpart = (tid & 1) * 16;    // element offset 0 or 16

  for (int kk = 0; kk < Kd; kk += GT_K) {
    // ---- stage A tile ----
    {
      unsigned short* dst = sA + lrow * SSTR + lpart;
      if (a_is_fp32) {
        const float* src = (const float*)Aany + (size_t)(m0 + lrow) * Kd + kk + lpart;
        float f[16];
        #pragma unroll
        for (int i = 0; i < 4; ++i) {
          float4 t = ((const float4*)src)[i];
          f[4*i+0] = t.x; f[4*i+1] = t.y; f[4*i+2] = t.z; f[4*i+3] = t.w;
        }
        u32x4 o0, o1;
        o0.x = pk2bf(f[0],f[1]);   o0.y = pk2bf(f[2],f[3]);
        o0.z = pk2bf(f[4],f[5]);   o0.w = pk2bf(f[6],f[7]);
        o1.x = pk2bf(f[8],f[9]);   o1.y = pk2bf(f[10],f[11]);
        o1.z = pk2bf(f[12],f[13]); o1.w = pk2bf(f[14],f[15]);
        *(u32x4*)dst       = o0;
        *(u32x4*)(dst + 8) = o1;
        if (kk + GT_K < Kd) __builtin_prefetch(src + GT_K, 0, 1);
      } else {
        const unsigned short* src =
            (const unsigned short*)Aany + (size_t)(m0 + lrow) * Kd + kk + lpart;
        async_g2l_b128(dst,     src);
        async_g2l_b128(dst + 8, src + 8);
      }
    }
    // ---- stage B tile (bf16, N-major over K) via async DMA ----
    {
      const unsigned short* src = Bt + (size_t)(n0 + lrow) * Kd + kk + lpart;
      unsigned short* dst = sB + lrow * SSTR + lpart;
      async_g2l_b128(dst,     src);
      async_g2l_b128(dst + 8, src + 8);
      if (kk + GT_K < Kd) __builtin_prefetch(src + GT_K, 0, 1);
    }
    wait_async0();
    __syncthreads();

    v16bf af[2], bfm[4];
    #pragma unroll
    for (int mt = 0; mt < 2; ++mt) af[mt]  = ld_frag(sA + (wm + mt*16) * SSTR, SSTR);
    #pragma unroll
    for (int nt = 0; nt < 4; ++nt) bfm[nt] = ld_frag(sB + (wn + nt*16) * SSTR, SSTR);
    #pragma unroll
    for (int mt = 0; mt < 2; ++mt)
      #pragma unroll
      for (int nt = 0; nt < 4; ++nt)
        acc[mt][nt] = wmma_bf16(af[mt], bfm[nt], acc[mt][nt]);
    __syncthreads();
  }

  // ---- epilogue: C layout lane holds column n=lane&15, rows r+8*(lane>>4) ----
  const int ncol  = lane & 15;
  const int nhalf = lane >> 4;
  #pragma unroll
  for (int nt = 0; nt < 4; ++nt) {
    const int n = n0 + wn + nt*16 + ncol;
    const float bvv = bias ? bias[n] : 0.0f;
    #pragma unroll
    for (int mt = 0; mt < 2; ++mt) {
      #pragma unroll
      for (int r = 0; r < 8; ++r) {
        const int m = m0 + wm + mt*16 + r + 8*nhalf;
        const float v = (acc[mt][nt][r] + bvv) * scale;
        if (mode == 0) {
          ((float*)Cout)[(size_t)m * N + n] = v;
        } else if (mode == 1) {
          const int hh = n >> 6, d = n & 63;
          ((unsigned short*)Cout)[((size_t)hh * S + m) * HDIM + d] = f2bf(v);
        } else {
          const int hh = n >> 6, d = n & 63;
          ((unsigned short*)Cout)[((size_t)hh * HDIM + d) * S + m] = f2bf(v);
        }
      }
    }
  }
}

// ---------------------------------------------------------------------------
// Fused flash attention, one (head, 128-query-block) per workgroup.
// q pre-scaled by 1/sqrt(d_model). Each wave owns 16 query rows; key loop in
// 64-key chunks staged into LDS with async DMA; online softmax in WMMA C-layout.
// ---------------------------------------------------------------------------
__global__ __launch_bounds__(256) void attn_fused(
    const unsigned short* __restrict__ qh,   // [H, S, 64] bf16 (scaled)
    const unsigned short* __restrict__ kh,   // [H, S, 64] bf16
    const unsigned short* __restrict__ vT,   // [H, 64, S] bf16
    unsigned short* __restrict__ attn,       // [S, 768] bf16 out
    int S)
{
  __shared__ __align__(16) unsigned short sK[64 * 72];      // keys x dims (padded)
  __shared__ __align__(16) unsigned short sV[64 * 72];      // dims x keys (padded)
  __shared__ __align__(16) unsigned short sP[8 * 16 * 64];  // per-wave P staging

  const int tid  = threadIdx.x;
  const int lane = tid & 31;
  const int w    = tid >> 5;
  const int h    = blockIdx.y;
  const int q0   = blockIdx.x * 128 + w * 16;   // this wave's first query row
  const int ncol = lane & 15;
  const int half = lane >> 4;

  // q fragments for this wave's 16 rows (K = 64 head dims -> 2 fragments)
  const unsigned short* qbase = qh + ((size_t)h * S + q0) * HDIM;
  const v16bf qf0 = ld_frag(qbase +  0, HDIM);
  const v16bf qf1 = ld_frag(qbase + 32, HDIM);

  const v8f vzero = {0.f,0.f,0.f,0.f,0.f,0.f,0.f,0.f};
  v8f oacc[4];
  #pragma unroll
  for (int dt = 0; dt < 4; ++dt) oacc[dt] = vzero;
  float mrun[8], lrun[8];
  #pragma unroll
  for (int r = 0; r < 8; ++r) { mrun[r] = -__builtin_inff(); lrun[r] = 0.0f; }

  const int crow  = tid >> 2;          // 0..63
  const int cpart = (tid & 3) * 16;    // 0/16/32/48
  const unsigned short* kgbase = kh + (size_t)h * S * HDIM;
  const unsigned short* vgbase = vT + (size_t)h * HDIM * S;
  unsigned short* pw = sP + w * (16 * 64);

  for (int kb = 0; kb < S; kb += 64) {
    // ---- async stage: 64 keys of K, 64 dims of V^T (ASYNCcnt-tracked) ----
    {
      const unsigned short* src = kgbase + (size_t)(kb + crow) * HDIM + cpart;
      unsigned short* dst = sK + crow * 72 + cpart;
      async_g2l_b128(dst,     src);
      async_g2l_b128(dst + 8, src + 8);
      src = vgbase + (size_t)crow * S + kb + cpart;
      dst = sV + crow * 72 + cpart;
      async_g2l_b128(dst,     src);
      async_g2l_b128(dst + 8, src + 8);
    }
    wait_async0();
    __syncthreads();

    // ---- S = q . k^T  (4 key-tiles of 16, K=64 -> 2 WMMA each) ----
    v8f st[4];
    #pragma unroll
    for (int nt = 0; nt < 4; ++nt) {
      v8f s = vzero;
      s = wmma_bf16(qf0, ld_frag(sK + nt*16*72 +  0, 72), s);
      s = wmma_bf16(qf1, ld_frag(sK + nt*16*72 + 32, 72), s);
      st[nt] = s;
    }

    // ---- online softmax per row (row = q0 + r + 8*half) ----
    #pragma unroll
    for (int r = 0; r < 8; ++r) {
      float mx = fmaxf(fmaxf(st[0][r], st[1][r]), fmaxf(st[2][r], st[3][r]));
      #pragma unroll
      for (int d = 1; d < 16; d <<= 1) mx = fmaxf(mx, __shfl_xor(mx, d));
      const float mnew  = fmaxf(mrun[r], mx);
      const float alpha = __expf(mrun[r] - mnew);
      mrun[r] = mnew;
      float rs = 0.0f;
      #pragma unroll
      for (int nt = 0; nt < 4; ++nt) {
        const float p = __expf(st[nt][r] - mnew);
        st[nt][r] = p;
        rs += p;
      }
      #pragma unroll
      for (int d = 1; d < 16; d <<= 1) rs += __shfl_xor(rs, d);
      lrun[r] = lrun[r] * alpha + rs;
      #pragma unroll
      for (int dt = 0; dt < 4; ++dt) oacc[dt][r] *= alpha;
    }

    // ---- relayout P (C-layout -> row-major) through per-wave LDS ----
    #pragma unroll
    for (int nt = 0; nt < 4; ++nt)
      #pragma unroll
      for (int r = 0; r < 8; ++r)
        pw[(r + 8*half) * 64 + nt*16 + ncol] = f2bf(st[nt][r]);
    // same-wave LDS RAW: DS ops are in-order, compiler inserts s_wait_dscnt
    const v16bf ap0 = ld_frag(pw +  0, 64);
    const v16bf ap1 = ld_frag(pw + 32, 64);

    // ---- O += P . V  (B lanes = output dims; rows of sV = dims over keys) ----
    #pragma unroll
    for (int dt = 0; dt < 4; ++dt) {
      oacc[dt] = wmma_bf16(ap0, ld_frag(sV + dt*16*72 +  0, 72), oacc[dt]);
      oacc[dt] = wmma_bf16(ap1, ld_frag(sV + dt*16*72 + 32, 72), oacc[dt]);
    }
    __syncthreads();   // protect sK/sV before next chunk
  }

  // ---- normalize + store bf16 into [S, 768] (heads concatenated) ----
  #pragma unroll
  for (int r = 0; r < 8; ++r) {
    const float inv = 1.0f / lrun[r];
    const int row = q0 + r + 8*half;
    #pragma unroll
    for (int dt = 0; dt < 4; ++dt)
      attn[(size_t)row * D_MODEL + h*HDIM + dt*16 + ncol] = f2bf(oacc[dt][r] * inv);
  }
}

// ---------------------------------------------------------------------------
extern "C" void kernel_launch(void* const* d_in, const int* in_sizes, int n_in,
                              void* d_out, int out_size, void* d_ws, size_t ws_size,
                              hipStream_t stream) {
  const float* Q  = (const float*)d_in[0];
  const float* Kx = (const float*)d_in[1];
  const float* V  = (const float*)d_in[2];
  const float* Wq = (const float*)d_in[3];
  const float* bq = (const float*)d_in[4];
  const float* Wk = (const float*)d_in[5];
  const float* bk = (const float*)d_in[6];
  const float* Wv = (const float*)d_in[7];
  const float* bv = (const float*)d_in[8];
  const float* Wo = (const float*)d_in[9];
  const float* bo = (const float*)d_in[10];
  const int S = in_sizes[0] / D_MODEL;   // 4096

  char* ws = (char*)d_ws;
  size_t off = 0;
  auto take = [&](size_t bytes) -> void* {
    void* p = ws + off;
    off = (off + bytes + 255) & ~(size_t)255;
    return p;
  };
  unsigned short* WqT = (unsigned short*)take((size_t)D_MODEL * D_MODEL * 2);
  unsigned short* WkT = (unsigned short*)take((size_t)D_MODEL * D_MODEL * 2);
  unsigned short* WvT = (unsigned short*)take((size_t)D_MODEL * D_MODEL * 2);
  unsigned short* WoT = (unsigned short*)take((size_t)D_MODEL * D_MODEL * 2);
  unsigned short* qh  = (unsigned short*)take((size_t)NHEADS * S * HDIM * 2);
  unsigned short* kh  = (unsigned short*)take((size_t)NHEADS * S * HDIM * 2);
  unsigned short* vTh = (unsigned short*)take((size_t)NHEADS * S * HDIM * 2);
  unsigned short* att = (unsigned short*)take((size_t)S * D_MODEL * 2);

  {
    const int n = D_MODEL * D_MODEL;
    dim3 g((n + 255) / 256), b(256);
    wt_cvt_t<<<g, b, 0, stream>>>(Wq, WqT, D_MODEL);
    wt_cvt_t<<<g, b, 0, stream>>>(Wk, WkT, D_MODEL);
    wt_cvt_t<<<g, b, 0, stream>>>(Wv, WvT, D_MODEL);
    wt_cvt_t<<<g, b, 0, stream>>>(Wo, WoT, D_MODEL);
  }

  const dim3 gg(S / GT_M, D_MODEL / GT_N), bb(256);
  const float qscale = 1.0f / sqrtf((float)D_MODEL);   // reference scales by sqrt(d_model)
  gemm_bf16_nt<<<gg, bb, 0, stream>>>(Q,  1, WqT, bq, qh,  S, D_MODEL, D_MODEL, 1, qscale, S);
  gemm_bf16_nt<<<gg, bb, 0, stream>>>(Kx, 1, WkT, bk, kh,  S, D_MODEL, D_MODEL, 1, 1.0f,  S);
  gemm_bf16_nt<<<gg, bb, 0, stream>>>(V,  1, WvT, bv, vTh, S, D_MODEL, D_MODEL, 2, 1.0f,  S);

  attn_fused<<<dim3(S / 128, NHEADS), 256, 0, stream>>>(qh, kh, vTh, att, S);

  gemm_bf16_nt<<<gg, bb, 0, stream>>>(att, 0, WoT, bo, d_out, S, D_MODEL, D_MODEL, 0, 1.0f, S);
}
